// LSTMSimple_49563922596352
// MI455X (gfx1250) — matrix operand order repaired and checked by reading, not compiled, not run on hardware
//
#include <hip/hip_runtime.h>
#include <math.h>

// ---------------------------------------------------------------------------
// LSTM(2 layers) + BN + dense head for gfx1250.
// Persistent per-layer kernel: recurrent+input weights staged in LDS (CDNA5's
// 320KB/WGP), all 512 timesteps executed in-kernel with a software grid
// barrier between GEMM (v_wmma bf16) and gate phases.
// ---------------------------------------------------------------------------

#define Bn 64
#define Tn 512
#define Fn 128
#define Hn 512
#define Gn 2048            // 4*H
#define BTn 32768          // B*T

typedef __attribute__((ext_vector_type(16))) __bf16          v16bf;
typedef __attribute__((ext_vector_type(8)))  float           v8f;
typedef __attribute__((ext_vector_type(8)))  unsigned short  u16x8;
typedef __attribute__((ext_vector_type(16))) unsigned short  u16x16;

static __device__ __forceinline__ unsigned short f2bf(float f) {
    unsigned int u = __float_as_uint(f);
    u = u + 0x7FFFu + ((u >> 16) & 1u);   // round-to-nearest-even
    return (unsigned short)(u >> 16);
}

static __device__ __forceinline__ float sigmoidf(float x) {
    return 1.0f / (1.0f + __expf(-x));
}

// ---- WMMA fragment loaders -------------------------------------------------
// A (16x32 bf16, row-major, base pre-offset to row m0):
//   lanes 0-15 : row r, K = k0+{0..15}
//   lanes 16-31: row r, K = k0+{16..31}
static __device__ __forceinline__ v16bf load_a_frag(const unsigned short* A,
                                                    size_t lda, int k0) {
    int lane = threadIdx.x & 31;
    int half = lane >> 4, r = lane & 15;
    const unsigned short* p = A + (size_t)r * lda + (size_t)(k0 + half * 16);
    u16x8 lo = *(const u16x8*)p;
    u16x8 hi = *(const u16x8*)(p + 8);
    u16x16 all = __builtin_shufflevector(lo, hi, 0,1,2,3,4,5,6,7,
                                                 8,9,10,11,12,13,14,15);
    return __builtin_bit_cast(v16bf, all);
}

// B (32x16 bf16) with B stored transposed BT[n][k]:
//   lanes 0-15 : col n=r, K = k0+{0..15}; lanes 16-31: K = k0+{16..31}
static __device__ __forceinline__ v16bf load_b_frag(const unsigned short* BTrow0,
                                                    size_t ldb, int k0) {
    int lane = threadIdx.x & 31;
    int half = lane >> 4, r = lane & 15;
    const unsigned short* p = BTrow0 + (size_t)r * ldb + (size_t)(k0 + half * 16);
    u16x16 all = *(const u16x16*)p;
    return __builtin_bit_cast(v16bf, all);
}

// Same but from an LDS-resident padded slice (ldb = Kt+8 elems -> 16B aligned,
// bank-rotating rows). Two b128 loads.
static __device__ __forceinline__ v16bf load_b_lds(const unsigned short* Blds,
                                                   int ldb, int n0, int k0) {
    int lane = threadIdx.x & 31;
    int half = lane >> 4, r = lane & 15;
    const unsigned short* p = Blds + (n0 + r) * ldb + k0 + half * 16;
    u16x8 lo = *(const u16x8*)p;
    u16x8 hi = *(const u16x8*)(p + 8);
    u16x16 all = __builtin_shufflevector(lo, hi, 0,1,2,3,4,5,6,7,
                                                 8,9,10,11,12,13,14,15);
    return __builtin_bit_cast(v16bf, all);
}

// ---- software grid barrier (generation counter, 0-init per call) ----------
static __device__ __forceinline__ void grid_sync(unsigned* bar) {
    __threadfence();
    __syncthreads();
    if (threadIdx.x == 0) {
        unsigned gen = atomicAdd(&bar[1], 0u);
        if (atomicAdd(&bar[0], 1u) == gridDim.x - 1) {
            atomicExch(&bar[0], 0u);
            atomicAdd(&bar[1], 1u);
        } else {
            while (atomicAdd(&bar[1], 0u) == gen) { __builtin_amdgcn_s_sleep(8); }
        }
    }
    __syncthreads();
    __threadfence();
}

// ---- Persistent LSTM layer -------------------------------------------------
// grid = 32 blocks, each owns 64 output columns of z; weights [64][Kt] staged
// into LDS once.  Per t: GEMM phase (z = [h|x_t]@W^T + b), grid barrier,
// gate phase (c,h update), grid barrier.
__global__ __launch_bounds__(256)
void lstm_layer(const unsigned short* __restrict__ BTg, int Kt,   // [2048][Kt]
                const unsigned short* __restrict__ A2base, size_t lda2, int K2,
                const float* __restrict__ bias,
                unsigned short* __restrict__ hb, float* __restrict__ cst,
                float* __restrict__ z, float* __restrict__ hs,
                unsigned* __restrict__ bar)
{
    __shared__ unsigned short wlds[64 * 1032];   // 129KB (max layer: Kt=1024,+8 pad)
    const int tid   = threadIdx.x;
    const int ncol0 = blockIdx.x * 64;
    const int Kp    = Kt + 8;                    // padded row stride (elems)

    // ---- stage weight slice into LDS (16B chunks) ----
    const int kv = Kt >> 3;
    for (int i = tid; i < 64 * kv; i += 256) {
        int rrow = i / kv;
        int ko   = (i - rrow * kv) * 8;
        u16x8 v = *(const u16x8*)(BTg + (size_t)(ncol0 + rrow) * (size_t)Kt + ko);
        *(u16x8*)(&wlds[rrow * Kp + ko]) = v;
    }
    __syncthreads();

    const int w    = tid >> 5;
    const int m0   = 16 * (w & 3);          // 4 m-tiles cover M=64
    const int ns   = (w >> 2) * 32;         // 2 n-groups x 2 subtiles cover N=64
    const int lane = tid & 31, half = lane >> 4, r = lane & 15;

    for (int t = 0; t < Tn; ++t) {
        // ---------------- GEMM phase ----------------
        v8f acc[2] = {};
        const unsigned short* A1m = hb + (size_t)m0 * Hn;
        for (int k0 = 0; k0 < Hn; k0 += 32) {
            v16bf a  = load_a_frag(A1m, Hn, k0);
            v16bf b0 = load_b_lds(wlds, Kp, ns,      k0);
            v16bf b1 = load_b_lds(wlds, Kp, ns + 16, k0);
            acc[0] = __builtin_amdgcn_wmma_f32_16x16x32_bf16(
                         false, a, false, b0, (short)0, acc[0], false, false);
            acc[1] = __builtin_amdgcn_wmma_f32_16x16x32_bf16(
                         false, a, false, b1, (short)0, acc[1], false, false);
        }
        const unsigned short* A2m =
            A2base + (size_t)t * (size_t)K2 + (size_t)m0 * lda2;
        for (int k0 = 0; k0 < K2; k0 += 32) {
            v16bf a  = load_a_frag(A2m, lda2, k0);
            v16bf b0 = load_b_lds(wlds, Kp, ns,      Hn + k0);
            v16bf b1 = load_b_lds(wlds, Kp, ns + 16, Hn + k0);
            acc[0] = __builtin_amdgcn_wmma_f32_16x16x32_bf16(
                         false, a, false, b0, (short)0, acc[0], false, false);
            acc[1] = __builtin_amdgcn_wmma_f32_16x16x32_bf16(
                         false, a, false, b1, (short)0, acc[1], false, false);
        }
#pragma unroll
        for (int j = 0; j < 2; ++j) {
            int col  = ncol0 + ns + 16 * j + r;
            float bv = bias[col];
#pragma unroll
            for (int v = 0; v < 8; ++v) {
                int row = m0 + v + 8 * half;
                z[(size_t)row * Gn + col] = acc[j][v] + bv;
            }
        }
        grid_sync(bar);
        // ---------------- gate phase ----------------
        for (int e = blockIdx.x * 256 + tid; e < Bn * Hn; e += 32 * 256) {
            int b = e >> 9, j = e & 511;
            const float* zr = z + (size_t)b * Gn;
            float zi = zr[j], zf = zr[Hn + j];
            float zg = zr[2 * Hn + j], zo = zr[3 * Hn + j];
            float cv = cst[e];
            cv = sigmoidf(zf) * cv + sigmoidf(zi) * tanhf(zg);
            float hv = sigmoidf(zo) * tanhf(cv);
            cst[e] = cv;
            hb[e]  = f2bf(hv);
            hs[((size_t)b * Tn + t) * Hn + j] = hv;
        }
        grid_sync(bar);
    }
}

// ---- BatchNorm: partial sums (atomic) then apply --------------------------
__global__ __launch_bounds__(256)
void bn_partial(const float* __restrict__ hs, float* __restrict__ acc)
{
    int tx = threadIdx.x & 63, ty = threadIdx.x >> 6;
    int j   = blockIdx.x * 64 + tx;
    int bt0 = blockIdx.y * 512;
    float s = 0.f, s2 = 0.f;
    for (int bt = bt0 + ty; bt < bt0 + 512; bt += 4) {
        float v = hs[(size_t)bt * Hn + j];
        s += v; s2 += v * v;
    }
    __shared__ float sm[8][64];
    sm[ty][tx] = s; sm[4 + ty][tx] = s2;
    __syncthreads();
    if (ty == 0) {
        s  = sm[0][tx] + sm[1][tx] + sm[2][tx] + sm[3][tx];
        s2 = sm[4][tx] + sm[5][tx] + sm[6][tx] + sm[7][tx];
        atomicAdd(&acc[j], s);
        atomicAdd(&acc[Hn + j], s2);
    }
}

__global__ __launch_bounds__(256)
void bn_apply(const float* __restrict__ hs, const float* __restrict__ acc,
              const float* __restrict__ gamma, const float* __restrict__ beta,
              unsigned short* __restrict__ outb)
{
    int idx = blockIdx.x * 256 + threadIdx.x;
    int j = idx & 511;
    const float invN = 1.0f / (float)BTn;
    float mean = acc[j] * invN;
    float var  = acc[Hn + j] * invN - mean * mean;
    float sc   = rsqrtf(var + 1e-5f);
    float v = (hs[idx] - mean) * sc * gamma[j] + beta[j];
    outb[idx] = f2bf(v);
}

// ---- Dense head: out = relu(h @ Wd1 + bd1) @ Wd2 + bd2 --------------------
__global__ __launch_bounds__(256)
void dense_head(const unsigned short* __restrict__ A,
                const unsigned short* __restrict__ BT,
                const float* __restrict__ bd1, const float* __restrict__ wd2,
                const float* __restrict__ bd2, float* __restrict__ out)
{
    int w = threadIdx.x >> 5;
    size_t m0 = (size_t)blockIdx.x * 128 + (size_t)w * 16;
    const unsigned short* Am = A + m0 * Hn;
    v8f acc = {};
    for (int k0 = 0; k0 < Hn; k0 += 32) {
        v16bf a = load_a_frag(Am, Hn, k0);
        v16bf b = load_b_frag(BT, Hn, k0);
        acc = __builtin_amdgcn_wmma_f32_16x16x32_bf16(
                  false, a, false, b, (short)0, acc, false, false);
    }
    int lane = threadIdx.x & 31, half = lane >> 4, r = lane & 15;
    float b1v = bd1[r];
    float w2v = wd2[r];
#pragma unroll
    for (int v = 0; v < 8; ++v) {
        float t = acc[v] + b1v;
        t = t > 0.f ? t : 0.f;
        t *= w2v;
        t += __shfl_xor(t, 1, 32);
        t += __shfl_xor(t, 2, 32);
        t += __shfl_xor(t, 4, 32);
        t += __shfl_xor(t, 8, 32);   // masks <=8 stay inside 16-lane half
        if (r == 0) out[m0 + v + 8 * half] = t + bd2[0];
    }
}

// ---- Prep kernels ----------------------------------------------------------
__global__ __launch_bounds__(256)
void f32_to_bf16_k(const float* __restrict__ s, unsigned short* __restrict__ d, int n)
{
    int i = blockIdx.x * 256 + threadIdx.x;
    if (i < n) d[i] = f2bf(s[i]);
}

// d[n][k] = (k<Kh ? Wh[k][n] : Wx[k-Kh][n]) as bf16
__global__ __launch_bounds__(256)
void pack_wT(const float* __restrict__ Wh, const float* __restrict__ Wx,
             unsigned short* __restrict__ d, int Kh, int Kt, int N, int total)
{
    int i = blockIdx.x * 256 + threadIdx.x;
    if (i >= total) return;
    int n = i / Kt, k = i - n * Kt;
    float v = (k < Kh) ? Wh[(size_t)k * N + n] : Wx[(size_t)(k - Kh) * N + n];
    d[i] = f2bf(v);
}

// d[n][k] = W[k][n]
__global__ __launch_bounds__(256)
void packT(const float* __restrict__ W, unsigned short* __restrict__ d,
           int K, int N, int total)
{
    int i = blockIdx.x * 256 + threadIdx.x;
    if (i >= total) return;
    int n = i / K, k = i - n * K;
    d[i] = f2bf(W[(size_t)k * N + n]);
}

__global__ __launch_bounds__(256)
void zero_f32(float* __restrict__ p, int n)
{
    int i = blockIdx.x * 256 + threadIdx.x;
    if (i < n) p[i] = 0.f;
}

// ---------------------------------------------------------------------------
extern "C" void kernel_launch(void* const* d_in, const int* in_sizes, int n_in,
                              void* d_out, int out_size, void* d_ws, size_t ws_size,
                              hipStream_t stream)
{
    (void)in_sizes; (void)n_in; (void)out_size; (void)ws_size;
    const float* x   = (const float*)d_in[0];
    const float* Wx1 = (const float*)d_in[1];
    const float* Wh1 = (const float*)d_in[2];
    const float* b1  = (const float*)d_in[3];
    const float* g1  = (const float*)d_in[4];
    const float* be1 = (const float*)d_in[5];
    const float* Wx2 = (const float*)d_in[6];
    const float* Wh2 = (const float*)d_in[7];
    const float* b2  = (const float*)d_in[8];
    const float* g2  = (const float*)d_in[9];
    const float* be2 = (const float*)d_in[10];
    const float* Wd1 = (const float*)d_in[11];
    const float* bd1 = (const float*)d_in[12];
    const float* Wd2 = (const float*)d_in[13];
    const float* bd2 = (const float*)d_in[14];
    float* out = (float*)d_out;

    // ---- workspace layout (256B aligned slabs) ----
    char* ws = (char*)d_ws;
    size_t off = 0;
    auto alloc = [&](size_t bytes) -> char* {
        char* p = ws + off;
        off += (bytes + 255) & ~(size_t)255;
        return p;
    };
    unsigned short* xb  = (unsigned short*)alloc((size_t)Bn*Tn*Fn*2);   // x bf16
    unsigned short* w1c = (unsigned short*)alloc((size_t)Gn*(Hn+Fn)*2); // [2048][640]
    unsigned short* w2c = (unsigned short*)alloc((size_t)Gn*(Hn+Hn)*2); // [2048][1024]
    unsigned short* wdt = (unsigned short*)alloc((size_t)16*Hn*2);      // Wd1^T
    float*          z   = (float*)alloc((size_t)Bn*Gn*4);               // step gates
    float*          hs  = (float*)alloc((size_t)BTn*Hn*4);  // raw h (shared L1/L2)
    unsigned short* h1n = (unsigned short*)alloc((size_t)BTn*Hn*2);     // BN1 out
    unsigned short* h2n = (unsigned short*)alloc((size_t)BTn*Hn*2);     // BN2 out
    // contiguous zero-initialized state block:
    float*          c1  = (float*)alloc((size_t)Bn*Hn*4);               // 128KB
    float*          c2  = (float*)alloc((size_t)Bn*Hn*4);               // 128KB
    unsigned short* h1b = (unsigned short*)alloc((size_t)Bn*Hn*2);      // 64KB
    unsigned short* h2b = (unsigned short*)alloc((size_t)Bn*Hn*2);      // 64KB
    float*          ac1 = (float*)alloc((size_t)2*Hn*4);                // 4KB
    float*          ac2 = (float*)alloc((size_t)2*Hn*4);                // 4KB
    unsigned*       bar = (unsigned*)alloc(256);                        // barrier

    // ---- zero state (c1..bar contiguous: 401664 bytes = 100416 floats) ----
    {
        int n = 100416;
        zero_f32<<<(n + 255) / 256, 256, 0, stream>>>(c1, n);
    }
    // ---- prep conversions ----
    {
        int n = Bn * Tn * Fn;
        f32_to_bf16_k<<<(n + 255) / 256, 256, 0, stream>>>(x, xb, n);
    }
    {
        int total = Gn * (Hn + Fn);
        pack_wT<<<(total + 255) / 256, 256, 0, stream>>>(Wh1, Wx1, w1c, Hn, Hn + Fn, Gn, total);
    }
    {
        int total = Gn * (Hn + Hn);
        pack_wT<<<(total + 255) / 256, 256, 0, stream>>>(Wh2, Wx2, w2c, Hn, Hn + Hn, Gn, total);
    }
    {
        int total = 16 * Hn;
        packT<<<(total + 255) / 256, 256, 0, stream>>>(Wd1, wdt, Hn, 16, total);
    }

    // ---- layer 1 (persistent): z = [h1 | x_t] @ W1c^T + b1 ----
    lstm_layer<<<32, 256, 0, stream>>>(w1c, Hn + Fn,
                                       xb, (size_t)Tn * Fn, Fn,
                                       b1, h1b, c1, z, hs, bar);
    // ---- BN1 ----
    bn_partial<<<dim3(Hn / 64, 64), 256, 0, stream>>>(hs, ac1);
    bn_apply<<<((size_t)BTn * Hn) / 256, 256, 0, stream>>>(hs, ac1, g1, be1, h1n);

    // ---- layer 2 (persistent): z = [h2 | h1n_t] @ W2c^T + b2 ----
    lstm_layer<<<32, 256, 0, stream>>>(w2c, Hn + Hn,
                                       h1n, (size_t)Tn * Hn, Hn,
                                       b2, h2b, c2, z, hs, bar);
    // ---- BN2 ----
    bn_partial<<<dim3(Hn / 64, 64), 256, 0, stream>>>(hs, ac2);
    bn_apply<<<((size_t)BTn * Hn) / 256, 256, 0, stream>>>(hs, ac2, g2, be2, h2n);

    // ---- dense head ----
    dense_head<<<BTn / 128, 256, 0, stream>>>(h2n, wdt, bd1, Wd2, bd2, out);
}
